// SpikingAttention_40956808135048
// MI455X (gfx1250) — matrix operand order repaired and checked by reading, not compiled, not run on hardware
//
#include <hip/hip_runtime.h>
#include <hip/hip_bf16.h>

// ---------------------------------------------------------------- constants
#define ALPHA_C 0.9f
#define RHO_C   0.995f
#define BETA_C  1.6f
#define B0_C    1.0f

constexpr int Bk = 32;        // batch
constexpr int Tk = 1000;      // timesteps
constexpr int Dk = 512;       // input dim
constexpr int Uk = 512;       // units
constexpr int BT = Bk * Tk;   // 32000 rows

typedef __attribute__((ext_vector_type(16))) _Float16 v16h;
typedef __attribute__((ext_vector_type(8)))  _Float16 v8h;
typedef __attribute__((ext_vector_type(8)))  float    v8f;

#define SHUF16(lo, hi) __builtin_shufflevector(lo, hi, 0,1,2,3,4,5,6,7,8,9,10,11,12,13,14,15)

__device__ inline v8f wmma_f16(v16h a, v16h b, v8f c) {
    return __builtin_amdgcn_wmma_f32_16x16x32_f16(false, a, false, b, (short)0, c, false, false);
}

// ---------------------------------------------------------------- conversion
__global__ void cvt_half_kernel(const float* __restrict__ s, _Float16* __restrict__ d, long n) {
    long i   = (long)blockIdx.x * blockDim.x + threadIdx.x;
    long str = (long)gridDim.x * blockDim.x;
    for (; i < n; i += str) d[i] = (_Float16)s[i];
}

// Wt[c*R + r] = W[r*C + c]  (transposed f16 weight; Wt rows indexed by output unit)
__global__ void transpose_half_kernel(const float* __restrict__ W, _Float16* __restrict__ Wt,
                                      int R, int C) {
    int i = blockIdx.x * blockDim.x + threadIdx.x;
    if (i < R * C) {
        int r = i / C, c = i % C;
        Wt[(long)c * R + r] = (_Float16)W[i];
    }
}

// ---------------------------------------------------------------- GEMM panel
// One 16x64 output tile per wave: A fragment shared across 4 WMMAs.
// ap: per-lane A pointer (row tm*16+(lane&15), + (lane>>4)*16 into K), unit stride in K.
// bp: per-lane B pointer (row tn*64+(lane&15) of Wt, + (lane>>4)*16 into K); subtiles at
//     +16*ldb, +32*ldb, +48*ldb rows (immediate offsets).
__device__ inline void gemm_panel4(const _Float16* __restrict__ ap,
                                   const _Float16* __restrict__ bp, long ldb, int kc,
                                   v8f& c0, v8f& c1, v8f& c2, v8f& c3) {
#pragma unroll 4
    for (int k = 0; k < kc; k += 32) {
        v8h lo = *(const v8h*)(ap + k);
        v8h hi = *(const v8h*)(ap + k + 8);
        v16h a  = SHUF16(lo, hi);
        v16h b0 = *(const v16h*)(bp + k);
        v16h b1 = *(const v16h*)(bp + 16 * ldb + k);
        v16h b2 = *(const v16h*)(bp + 32 * ldb + k);
        v16h b3 = *(const v16h*)(bp + 48 * ldb + k);
        c0 = wmma_f16(a, b0, c0);
        c1 = wmma_f16(a, b1, c1);
        c2 = wmma_f16(a, b2, c2);
        c3 = wmma_f16(a, b3, c3);
    }
}

// C[M,N] = [A1 | A2][M,K] * Wt^T  (A split at ksplit; pass ksplit=K for no split)
__global__ __launch_bounds__(256) void gemm_ffn_kernel(
        const _Float16* __restrict__ A1, const _Float16* __restrict__ A2, int ksplit,
        const _Float16* __restrict__ Wt, float* __restrict__ C, int M, int K, int N) {
    const int wave = (int)((blockIdx.x * 256 + threadIdx.x) >> 5);
    const int lane = threadIdx.x & 31;
    const int tiles_n = N >> 6;                 // 64-wide N tiles
    const int tm = wave / tiles_n, tn = wave % tiles_n;
    if (tm * 16 >= M) return;

    const int lh = lane >> 4, ll = lane & 15;
    v8f c0 = {}, c1 = {}, c2 = {}, c3 = {};

    {   // panel over A1, K range [0, ksplit)
        const _Float16* ap = A1 + (long)(tm * 16 + ll) * ksplit + (lh << 4);
        const _Float16* bp = Wt + (long)(tn * 64 + ll) * K + (lh << 4);
        gemm_panel4(ap, bp, K, ksplit, c0, c1, c2, c3);
    }
    if (ksplit < K) {   // panel over A2, K range [ksplit, K)
        const int kc2 = K - ksplit;
        const _Float16* ap = A2 + (long)(tm * 16 + ll) * kc2 + (lh << 4);
        const _Float16* bp = Wt + (long)(tn * 64 + ll) * K + ksplit + (lh << 4);
        gemm_panel4(ap, bp, K, kc2, c0, c1, c2, c3);
    }
    // store 4 tiles: element r -> row mbase+r, col tn*64 + s*16 + ll
    const int mbase = tm * 16 + (lh << 3);
    float* cp = C + (long)mbase * N + tn * 64 + ll;
#pragma unroll
    for (int r = 0; r < 8; r++) {
        cp[(long)r * N + 0]  = c0[r];
        cp[(long)r * N + 16] = c1[r];
        cp[(long)r * N + 32] = c2[r];
        cp[(long)r * N + 48] = c3[r];
    }
}

// ---------------------------------------------------------------- LSNN scan
// One workgroup (1024 thr / 32 waves) runs the whole T-step scan for one module.
template <bool HAS_ZG, bool HAS_OUT0>
__device__ inline void lsnn_scan_body(const float* __restrict__ I,
                                      const _Float16* __restrict__ Wrec_t,
                                      _Float16* __restrict__ zg, float* __restrict__ out0,
                                      float* __restrict__ out1, float* __restrict__ out2,
                                      float* __restrict__ out3,
                                      int slot_sp, int slot_thr, int slot_vsc,
                                      _Float16* zsh, float* cursh) {
    const int tid  = threadIdx.x;
    const int lane = tid & 31;
    const int wave = tid >> 5;
    const int b    = wave;                  // wave w owns batch row b=w elementwise
    const int u0   = (tid & 31) << 4;       // 16 contiguous units per thread
    const int lh = lane >> 4, ll = lane & 15;

    float v[16], a[16], zp[16];
#pragma unroll
    for (int j = 0; j < 16; j++) { v[j] = 0.f; a[j] = 0.f; zp[j] = 0.f; }
    for (int i = tid; i < Bk * Uk; i += 1024) zsh[i] = (_Float16)0.0f;

    // t-invariant GEMM lane pointers
    const int n0 = wave << 4;
    const _Float16* ap0 = zsh + ll * Uk + (lh << 4);        // rows 0..15 of z
    const _Float16* ap1 = ap0 + 16 * Uk;                    // rows 16..31
    const _Float16* bp  = Wrec_t + (long)(n0 + ll) * Uk + (lh << 4);
    float* cw0 = cursh + (lh << 3) * Uk + n0 + ll;
    float* cw1 = cw0 + 16 * Uk;

    // t-invariant elementwise pointers (bumped per step)
    _Float16*    zwr = zsh + b * Uk + u0;
    const float* crd = cursh + b * Uk + u0;
    const float* Ip  = I + (long)b * Tk * Uk + u0;
    _Float16* zgp = HAS_ZG   ? zg   + (long)b * Tk * Uk + u0 : (_Float16*)nullptr;
    float*    o0  = HAS_OUT0 ? out0 + (long)b * Tk * Uk + u0 : (float*)nullptr;
    float* o1 = out1 + (long)b * Tk * (6 * Uk) + slot_sp  * Uk + u0;
    float* o2 = out2 + (long)b * Tk * (4 * Uk) + slot_thr * Uk + u0;
    float* o3 = out3 + (long)b * Tk * (6 * Uk) + slot_vsc * Uk + u0;

    for (int t = 0; t < Tk; t++) {
        __syncthreads();                    // z_prev (or init) visible in zsh
        // ---- recurrent GEMM: cur[32,512] = z_prev @ Wrec ----
        v8f c0 = {}, c1 = {};
#pragma unroll
        for (int k = 0; k < Uk; k += 32) {
            v8h lo0 = *(const v8h*)(ap0 + k), hi0 = *(const v8h*)(ap0 + k + 8);
            v8h lo1 = *(const v8h*)(ap1 + k), hi1 = *(const v8h*)(ap1 + k + 8);
            v16h a0 = SHUF16(lo0, hi0);
            v16h a1 = SHUF16(lo1, hi1);
            v16h bf = *(const v16h*)(bp + k);
            c0 = wmma_f16(a0, bf, c0);
            c1 = wmma_f16(a1, bf, c1);
        }
#pragma unroll
        for (int r = 0; r < 8; r++) { cw0[r * Uk] = c0[r]; cw1[r * Uk] = c1[r]; }
        __syncthreads();
        // ---- adaptive-LIF elementwise update ----
        v8h zlo, zhi;
#pragma unroll
        for (int j = 0; j < 16; j++) {
            float cur = crd[j] + Ip[j];
            float an  = RHO_C * a[j] + zp[j];
            float thr = B0_C + BETA_C * an;
            float vn  = ALPHA_C * v[j] + cur - zp[j] * thr;
            float vsc = (vn - thr) * (1.0f / B0_C);
            float z   = vsc > 0.f ? 1.f : 0.f;
            v[j] = vn; a[j] = an; zp[j] = z;
            if (j < 8) zlo[j] = (_Float16)z; else zhi[j - 8] = (_Float16)z;
            if (HAS_OUT0) o0[j] = z;
            o1[j] = z; o2[j] = thr; o3[j] = vsc;
        }
        *(v8h*)zwr = zlo; *(v8h*)(zwr + 8) = zhi;
        if (HAS_ZG) { *(v8h*)zgp = zlo; *(v8h*)(zgp + 8) = zhi; zgp += Uk; }
        if (HAS_OUT0) o0 += Uk;
        Ip += Uk; o1 += 6 * Uk; o2 += 4 * Uk; o3 += 6 * Uk;
    }
}

__global__ __launch_bounds__(1024) void lsnn3_scan_kernel(
        const float* Ik, const float* Iq, const float* Iv,
        const _Float16* Wk, const _Float16* Wq, const _Float16* Wv,
        _Float16* zk, _Float16* zq, _Float16* zv,
        float* out1, float* out2, float* out3) {
    __shared__ __attribute__((aligned(32))) _Float16 zsh[Bk * Uk];
    __shared__ float cursh[Bk * Uk];
    int m = blockIdx.x;
    const float*    I = (m == 0) ? Ik : (m == 1) ? Iq : Iv;
    const _Float16* W = (m == 0) ? Wk : (m == 1) ? Wq : Wv;
    _Float16*      zg = (m == 0) ? zk : (m == 1) ? zq : zv;
    lsnn_scan_body<true, false>(I, W, zg, nullptr, out1, out2, out3, m, m, m, zsh, cursh);
}

__global__ __launch_bounds__(1024) void lsnn_att_scan_kernel(
        const float* I, const _Float16* W,
        float* out0, float* out1, float* out2, float* out3) {
    __shared__ __attribute__((aligned(32))) _Float16 zsh[Bk * Uk];
    __shared__ float cursh[Bk * Uk];
    lsnn_scan_body<false, true>(I, W, nullptr, out0, out1, out2, out3, 5, 3, 5, zsh, cursh);
}

// ---------------------------------------------------------------- WTA scan
__global__ __launch_bounds__(1024) void wta_scan_kernel(
        const float* __restrict__ I, const _Float16* __restrict__ Wb_t,
        _Float16* __restrict__ zAg, float* __restrict__ out1, float* __restrict__ out3) {
    __shared__ __attribute__((aligned(32))) _Float16 zsh[Bk * Uk];
    __shared__ float cursh[Bk * Uk];
    __shared__ float red[Bk][32];

    const int tid  = threadIdx.x;
    const int lane = tid & 31;
    const int wave = tid >> 5;
    const int b    = wave;
    const int u0   = (tid & 31) << 4;
    const int lh = lane >> 4, ll = lane & 15;

    float vA[16], vB[16], zA[16], zB[16];
#pragma unroll
    for (int j = 0; j < 16; j++) { vA[j] = 0.f; vB[j] = 0.f; zA[j] = 0.f; zB[j] = 0.f; }

    const int n0 = wave << 4;
    const _Float16* ap0 = zsh + ll * Uk + (lh << 4);
    const _Float16* ap1 = ap0 + 16 * Uk;
    const _Float16* bp  = Wb_t + (long)(n0 + ll) * Uk + (lh << 4);
    float* cw0 = cursh + (lh << 3) * Uk + n0 + ll;
    float* cw1 = cw0 + 16 * Uk;

    _Float16*    zwr = zsh + b * Uk + u0;
    const float* crd = cursh + b * Uk + u0;
    const float* Ip  = I + (long)b * Tk * Uk + u0;
    _Float16* zgp = zAg + (long)b * Tk * Uk + u0;
    float* o1a = out1 + (long)b * Tk * (6 * Uk) + 3 * Uk + u0;
    float* o3a = out3 + (long)b * Tk * (6 * Uk) + 3 * Uk + u0;
    float* o1b = o1a + Uk;   // slot 4
    float* o3b = o3a + Uk;   // slot 4

    for (int t = 0; t < Tk; t++) {
        // ---- pop A membrane update + per-row max ----
        float vAn[16];
        float lmax = -1e30f;
#pragma unroll
        for (int j = 0; j < 16; j++) {
            vAn[j] = ALPHA_C * vA[j] + Ip[j] - zA[j] * B0_C;
            lmax = fmaxf(lmax, vAn[j]);
        }
        red[b][lane] = lmax;
        __syncthreads();
        float rmax = -1e30f;
#pragma unroll
        for (int g = 0; g < 32; g++) rmax = fmaxf(rmax, red[b][g]);
        // ---- winner-take-all spike ----
        v8h zlo, zhi;
#pragma unroll
        for (int j = 0; j < 16; j++) {
            float vsc = (vAn[j] - B0_C) * (1.0f / B0_C);
            float win = (vAn[j] == rmax) ? 1.f : 0.f;
            float z   = (vsc > 0.f ? 1.f : 0.f) * win;
            vA[j] = vAn[j]; zA[j] = z;
            if (j < 8) zlo[j] = (_Float16)z; else zhi[j - 8] = (_Float16)z;
            o1a[j] = z; o3a[j] = vsc;
        }
        *(v8h*)zwr = zlo; *(v8h*)(zwr + 8) = zhi;
        *(v8h*)zgp = zlo; *(v8h*)(zgp + 8) = zhi;
        __syncthreads();
        // ---- curB = zA_new @ Wb ----
        v8f c0 = {}, c1 = {};
#pragma unroll
        for (int k = 0; k < Uk; k += 32) {
            v8h lo0 = *(const v8h*)(ap0 + k), hi0 = *(const v8h*)(ap0 + k + 8);
            v8h lo1 = *(const v8h*)(ap1 + k), hi1 = *(const v8h*)(ap1 + k + 8);
            v16h a0 = SHUF16(lo0, hi0);
            v16h a1 = SHUF16(lo1, hi1);
            v16h bf = *(const v16h*)(bp + k);
            c0 = wmma_f16(a0, bf, c0);
            c1 = wmma_f16(a1, bf, c1);
        }
#pragma unroll
        for (int r = 0; r < 8; r++) { cw0[r * Uk] = c0[r]; cw1[r * Uk] = c1[r]; }
        __syncthreads();
        // ---- pop B update ----
#pragma unroll
        for (int j = 0; j < 16; j++) {
            float vBn  = ALPHA_C * vB[j] + crd[j] - zB[j] * B0_C;
            float vscB = (vBn - B0_C) * (1.0f / B0_C);
            float z    = vscB > 0.f ? 1.f : 0.f;
            vB[j] = vBn; zB[j] = z;
            o1b[j] = z; o3b[j] = vscB;
        }
        __syncthreads();   // protect red/zsh for next step
        Ip += Uk; zgp += Uk;
        o1a += 6 * Uk; o3a += 6 * Uk; o1b += 6 * Uk; o3b += 6 * Uk;
    }
}

// ---------------------------------------------------------------- host
extern "C" void kernel_launch(void* const* d_in, const int* in_sizes, int n_in,
                              void* d_out, int out_size, void* d_ws, size_t ws_size,
                              hipStream_t stream) {
    const float* x        = (const float*)d_in[0];
    const float* Wk_in    = (const float*)d_in[1];
    const float* Wk_rec   = (const float*)d_in[2];
    const float* Wq_in    = (const float*)d_in[3];
    const float* Wq_rec   = (const float*)d_in[4];
    const float* Wv_in    = (const float*)d_in[5];
    const float* Wv_rec   = (const float*)d_in[6];
    const float* Watt_in  = (const float*)d_in[7];
    const float* Watt_rec = (const float*)d_in[8];
    const float* Wwta_a   = (const float*)d_in[9];
    const float* Wwta_b   = (const float*)d_in[10];

    float* out = (float*)d_out;
    const long BTU = (long)BT * Uk;               // 16,384,000
    float* out0 = out;                            // attention spikes [B,T,U]
    float* out1 = out + BTU;                      // all_spikes [B,T,6U]
    float* out2 = out + 7 * BTU;                  // all_thr    [B,T,4U]
    float* out3 = out + 11 * BTU;                 // all_v_sc   [B,T,6U]

    // ---- workspace carve-out ----
    char* ws = (char*)d_ws;
    size_t off = 0;
    auto carve = [&](size_t bytes) -> char* {
        char* p = ws + off;
        off += (bytes + 255) & ~(size_t)255;
        return p;
    };
    _Float16* xh        = (_Float16*)carve((size_t)BT * Dk * 2);
    _Float16* wkin_t    = (_Float16*)carve((size_t)Dk * Uk * 2);
    _Float16* wqin_t    = (_Float16*)carve((size_t)Dk * Uk * 2);
    _Float16* wvin_t    = (_Float16*)carve((size_t)Dk * Uk * 2);
    _Float16* wkrec_t   = (_Float16*)carve((size_t)Uk * Uk * 2);
    _Float16* wqrec_t   = (_Float16*)carve((size_t)Uk * Uk * 2);
    _Float16* wvrec_t   = (_Float16*)carve((size_t)Uk * Uk * 2);
    _Float16* wattrec_t = (_Float16*)carve((size_t)Uk * Uk * 2);
    _Float16* wwtab_t   = (_Float16*)carve((size_t)Uk * Uk * 2);
    _Float16* wattin_t  = (_Float16*)carve((size_t)2 * Uk * Uk * 2);
    _Float16* wwtaa_t   = (_Float16*)carve((size_t)2 * Uk * Uk * 2);
    float*    IA        = (float*)carve((size_t)BT * Uk * 4);   // I_k, later I_wta
    float*    IB        = (float*)carve((size_t)BT * Uk * 4);   // I_q, later I_att
    float*    IC        = (float*)carve((size_t)BT * Uk * 4);   // I_v
    _Float16* zk        = (_Float16*)carve((size_t)BT * Uk * 2);
    _Float16* zq        = (_Float16*)carve((size_t)BT * Uk * 2);
    _Float16* zv        = (_Float16*)carve((size_t)BT * Uk * 2);
    _Float16* zA        = (_Float16*)carve((size_t)BT * Uk * 2);
    (void)ws_size; (void)in_sizes; (void)n_in; (void)out_size;

    // ---- phase 0: precision conversion + weight transpose ----
    cvt_half_kernel<<<4096, 256, 0, stream>>>(x, xh, (long)BT * Dk);
    const int tW1 = (Dk * Uk + 255) / 256;        // 512x512
    const int tW2 = (2 * Uk * Uk + 255) / 256;    // 1024x512
    transpose_half_kernel<<<tW1, 256, 0, stream>>>(Wk_in,    wkin_t,    Dk,     Uk);
    transpose_half_kernel<<<tW1, 256, 0, stream>>>(Wq_in,    wqin_t,    Dk,     Uk);
    transpose_half_kernel<<<tW1, 256, 0, stream>>>(Wv_in,    wvin_t,    Dk,     Uk);
    transpose_half_kernel<<<tW1, 256, 0, stream>>>(Wk_rec,   wkrec_t,   Uk,     Uk);
    transpose_half_kernel<<<tW1, 256, 0, stream>>>(Wq_rec,   wqrec_t,   Uk,     Uk);
    transpose_half_kernel<<<tW1, 256, 0, stream>>>(Wv_rec,   wvrec_t,   Uk,     Uk);
    transpose_half_kernel<<<tW1, 256, 0, stream>>>(Watt_rec, wattrec_t, Uk,     Uk);
    transpose_half_kernel<<<tW1, 256, 0, stream>>>(Wwta_b,   wwtab_t,   Uk,     Uk);
    transpose_half_kernel<<<tW2, 256, 0, stream>>>(Watt_in,  wattin_t,  2 * Uk, Uk);
    transpose_half_kernel<<<tW2, 256, 0, stream>>>(Wwta_a,   wwtaa_t,   2 * Uk, Uk);

    // ---- phase 1: feed-forward currents I_k/I_q/I_v (WMMA GEMMs) ----
    const int gemm_blocks = ((BT / 16) * (Uk / 64) + 7) / 8;   // 2000 blocks x 8 waves
    gemm_ffn_kernel<<<gemm_blocks, 256, 0, stream>>>(xh, xh, Dk, wkin_t, IA, BT, Dk, Uk);
    gemm_ffn_kernel<<<gemm_blocks, 256, 0, stream>>>(xh, xh, Dk, wqin_t, IB, BT, Dk, Uk);
    gemm_ffn_kernel<<<gemm_blocks, 256, 0, stream>>>(xh, xh, Dk, wvin_t, IC, BT, Dk, Uk);

    // ---- phase 2: k/q/v adaptive-LIF scans (3 concurrent persistent WGs) ----
    lsnn3_scan_kernel<<<3, 1024, 0, stream>>>(IA, IB, IC,
                                              wkrec_t, wqrec_t, wvrec_t,
                                              zk, zq, zv, out1, out2, out3);

    // ---- phase 3: WTA input current I_wta = [zk|zq] @ Wwta_a (reuse IA) ----
    gemm_ffn_kernel<<<gemm_blocks, 256, 0, stream>>>(zk, zq, Uk, wwtaa_t, IA, BT, 2 * Uk, Uk);
    wta_scan_kernel<<<1, 1024, 0, stream>>>(IA, wwtab_t, zA, out1, out3);

    // ---- phase 4: attention current I_att = [zv|zA] @ Watt_in (reuse IB), then scan ----
    gemm_ffn_kernel<<<gemm_blocks, 256, 0, stream>>>(zv, zA, Uk, wattin_t, IB, BT, 2 * Uk, Uk);
    lsnn_att_scan_kernel<<<1, 1024, 0, stream>>>(IB, wattrec_t, out0, out1, out2, out3);
}
